// AVWDCRNN_47064251629612
// MI455X (gfx1250) — matrix operand order repaired and checked by reading, not compiled, not run on hardware
//
#include <hip/hip_runtime.h>
#include <hip/hip_bf16.h>
#include <math.h>

// Problem constants
#define Bq    32
#define Tq    24
#define Nq    2048
#define DOUTq 64
#define Cq    65          // DIN + DOUT (valid channels)
#define CP    80          // padded channels
#define KCP   160         // 2 * CP, K-dim of per-node matmul (5 chunks of 32)
#define XCOLP 2560        // Bq * CP, padded GEMM N dimension (80 tiles of 32)
#define OG    128         // gate output cols
#define OC    64          // candidate output cols
#define NKC   5           // KCP / 32

typedef __bf16 bf16_t;
typedef __attribute__((ext_vector_type(16))) __bf16 v16bf;
typedef __attribute__((ext_vector_type(8)))  float  v8f;
typedef __attribute__((ext_vector_type(4)))  unsigned int v4u;
typedef __attribute__((ext_vector_type(8)))  int v8i;
typedef __attribute__((ext_vector_type(4)))  int v4i;

__device__ __forceinline__ v8f wmma_bf16(v16bf a, v16bf b, v8f c) {
  return __builtin_amdgcn_wmma_f32_16x16x32_bf16(false, a, false, b, (short)0, c, false, false);
}

// Build a 16-element bf16 fragment from two 16-byte chunks (8 bf16 each).
__device__ __forceinline__ v16bf frag2q(const bf16_t* p0, const bf16_t* p1) {
  union { uint4 q[2]; v16bf v; } u;
  u.q[0] = *(const uint4*)p0;
  u.q[1] = *(const uint4*)p1;
  return u.v;
}

// A (16x32 bf16, row-major, ld): lane row = lane&15; runs [kb..kb+7],[kb+16..kb+23]
__device__ __forceinline__ v16bf frag_a_rm(const bf16_t* p, int ld) {
  int lane = threadIdx.x & 31;
  const bf16_t* r = p + (lane & 15) * ld + ((lane & 16) ? 8 : 0);
  return frag2q(r, r + 16);
}

// B (32x16 bf16) from column-major tile[col][k], leading dim ldk
__device__ __forceinline__ v16bf frag_b_cm(const bf16_t* p, int ldk) {
  int lane = threadIdx.x & 31;
  const bf16_t* r = p + (lane & 15) * ldk + ((lane & 16) ? 8 : 0);
  return frag2q(r, r + 16);
}

// Per-node activation A-fragment: K [0,80) from X, [80,160) from AX.
__device__ __forceinline__ v16bf frag_act(const bf16_t* __restrict__ X,
                                          const bf16_t* __restrict__ AX,
                                          int n, int mt, int kc) {
  int lane = threadIdx.x & 31;
  int b    = mt * 16 + (lane & 15);
  int kb   = (lane & 16) ? 8 : 0;
  size_t base = (size_t)n * XCOLP + (size_t)b * CP;
  int kg0 = kc * 32 + kb;
  int kg1 = kg0 + 16;
  const bf16_t* p0 = (kg0 >= CP) ? (AX + base + (kg0 - CP)) : (X + base + kg0);
  const bf16_t* p1 = (kg1 >= CP) ? (AX + base + (kg1 - CP)) : (X + base + kg1);
  return frag2q(p0, p1);
}

// ---- Kernel 1: adjacency softmax(relu(E E^T)) -> bf16 A[N][N] --------------
__global__ __launch_bounds__(256) void build_adj(const float* __restrict__ E,
                                                 bf16_t* __restrict__ A) {
  __shared__ float s[Nq];
  __shared__ float red[256];
  int n = blockIdx.x, tid = threadIdx.x;
  float e[10];
#pragma unroll
  for (int d = 0; d < 10; ++d) e[d] = E[n * 10 + d];

  float lmax = -1e30f;
  for (int j = tid; j < Nq; j += 256) {
    float dot = 0.f;
#pragma unroll
    for (int d = 0; d < 10; ++d) dot += e[d] * E[j * 10 + d];
    dot = fmaxf(dot, 0.f);
    s[j] = dot;
    lmax = fmaxf(lmax, dot);
  }
  red[tid] = lmax; __syncthreads();
  for (int st = 128; st > 0; st >>= 1) {
    if (tid < st) red[tid] = fmaxf(red[tid], red[tid + st]);
    __syncthreads();
  }
  float mx = red[0]; __syncthreads();

  float lsum = 0.f;
  for (int j = tid; j < Nq; j += 256) {
    float ex = __expf(s[j] - mx);
    s[j] = ex; lsum += ex;
  }
  red[tid] = lsum; __syncthreads();
  for (int st = 128; st > 0; st >>= 1) {
    if (tid < st) red[tid] += red[tid + st];
    __syncthreads();
  }
  float inv = 1.f / red[0]; __syncthreads();

  for (int j = tid; j < Nq; j += 256)
    A[(size_t)n * Nq + j] = (bf16_t)(s[j] * inv);
}

// ---- Kernel 2: per-node weights, pre-swizzled into WMMA B-tiles ------------
__global__ __launch_bounds__(256) void build_weights(
    const float* __restrict__ E,
    const float* __restrict__ Wg0, const float* __restrict__ bg0,
    const float* __restrict__ Wc0, const float* __restrict__ bc0,
    bf16_t* __restrict__ Wg, bf16_t* __restrict__ Wc,
    float* __restrict__ bg, float* __restrict__ bc) {
  int n = blockIdx.x, tid = threadIdx.x;
  __shared__ float e[10];
  if (tid < 10) e[tid] = E[n * 10 + tid];
  __syncthreads();

  for (int idx = tid; idx < NKC * 8 * 16 * 32; idx += 256) {
    int k_in = idx & 31;
    int colt = (idx >> 5) & 15;
    int ot   = (idx >> 9) & 7;
    int kc   = idx >> 12;
    int o    = ot * 16 + colt;
    int kg   = kc * 32 + k_in;
    int kk   = (kg >= CP) ? 1 : 0;
    int c    = kg - kk * CP;
    float v = 0.f;
    if (c < Cq) {
#pragma unroll
      for (int d = 0; d < 10; ++d)
        v += e[d] * Wg0[(((size_t)d * 2 + kk) * Cq + c) * OG + o];
    }
    Wg[(size_t)n * (NKC * 8 * 16 * 32) + idx] = (bf16_t)v;
  }
  for (int idx = tid; idx < NKC * 4 * 16 * 32; idx += 256) {
    int k_in = idx & 31;
    int colt = (idx >> 5) & 15;
    int ot   = (idx >> 9) & 3;
    int kc   = idx >> 11;
    int o    = ot * 16 + colt;
    int kg   = kc * 32 + k_in;
    int kk   = (kg >= CP) ? 1 : 0;
    int c    = kg - kk * CP;
    float v = 0.f;
    if (c < Cq) {
#pragma unroll
      for (int d = 0; d < 10; ++d)
        v += e[d] * Wc0[(((size_t)d * 2 + kk) * Cq + c) * OC + o];
    }
    Wc[(size_t)n * (NKC * 4 * 16 * 32) + idx] = (bf16_t)v;
  }
  for (int o = tid; o < OG; o += 256) {
    float v = 0.f;
#pragma unroll
    for (int d = 0; d < 10; ++d) v += e[d] * bg0[d * OG + o];
    bg[n * OG + o] = v;
  }
  for (int o = tid; o < OC; o += 256) {
    float v = 0.f;
#pragma unroll
    for (int d = 0; d < 10; ++d) v += e[d] * bc0[d * OC + o];
    bc[n * OC + o] = v;
  }
}

// ---- Pack kernels: X[m, b*80+c] (bf16, zero padded c>=65) ------------------
__global__ __launch_bounds__(256) void pack_gate(const float* __restrict__ x,
                                                 const float* __restrict__ h,
                                                 bf16_t* __restrict__ X, int t) {
  size_t i = (size_t)blockIdx.x * 256 + threadIdx.x;
  if (i >= (size_t)Nq * Bq * CP) return;
  int c = (int)(i % CP);
  size_t mb = i / CP;
  int b = (int)(mb % Bq);
  int m = (int)(mb / Bq);
  float v = 0.f;
  if (c == 0)      v = x[((size_t)b * Tq + t) * Nq + m];
  else if (c < Cq) v = h[((size_t)b * Nq + m) * DOUTq + (c - 1)];
  X[(size_t)m * XCOLP + (size_t)b * CP + c] = (bf16_t)v;
}

__global__ __launch_bounds__(256) void pack_cand(const float* __restrict__ x,
                                                 const float* __restrict__ h,
                                                 const float* __restrict__ r,
                                                 bf16_t* __restrict__ X, int t) {
  size_t i = (size_t)blockIdx.x * 256 + threadIdx.x;
  if (i >= (size_t)Nq * Bq * CP) return;
  int c = (int)(i % CP);
  size_t mb = i / CP;
  int b = (int)(mb % Bq);
  int m = (int)(mb / Bq);
  float v = 0.f;
  if (c == 0)      v = x[((size_t)b * Tq + t) * Nq + m];
  else if (c < Cq) {
    size_t idx = ((size_t)b * Nq + m) * DOUTq + (c - 1);
    v = r[idx] * h[idx];
  }
  X[(size_t)m * XCOLP + (size_t)b * CP + c] = (bf16_t)v;
}

// ---- GEMM: AX[n, col] = sum_m A[n,m] * X[m,col]  (2048 x 2560 x 2048) ------
// Block tile 256(M) x 32(N); 8 waves, each 32x32 (2x2 wmma tiles).
// A tile DMA'd global->LDS by the Tensor Data Mover (with LDS padding to
// stride 40), X tile staged transposed by the waves. Both double-buffered so
// TDM + global loads for chunk k+1 overlap the WMMAs of chunk k.
__global__ __launch_bounds__(256) void gemm_ax(const bf16_t* __restrict__ A,
                                               const bf16_t* __restrict__ X,
                                               bf16_t* __restrict__ AX) {
  __shared__ __align__(16) bf16_t la[2][256 * 40];   // [row][k], stride 40
  __shared__ __align__(16) bf16_t lxt[2][32 * 40];   // [col][k], stride 40
  int tid  = threadIdx.x;
  int wave = tid >> 5;
  int rowBase = blockIdx.x * 256;
  int colBase = blockIdx.y * 32;

  v8f z = {0.f, 0.f, 0.f, 0.f, 0.f, 0.f, 0.f, 0.f};
  v8f acc00 = z, acc01 = z, acc10 = z, acc11 = z;

  int xc  = tid & 31;   // local col for X staging
  int xkq = tid >> 5;   // k-quad 0..7 (4 k each)

  // TDM descriptor issue: 2D tile 256 rows x 32 bf16, tensor stride 2048,
  // LDS pad: 4 DWORDs after every 16 DWORDs -> LDS row stride 40 elements.
  auto issue_tdm = [&](int kc, int buf) {
    unsigned long long gaddr =
        (unsigned long long)(uintptr_t)(A + (size_t)rowBase * Nq + (size_t)kc * 32);
    unsigned int laddr = (unsigned int)(uintptr_t)(&la[buf][0]);
    v4u g0;
    g0.x = 1u;                                   // count=1, user mode
    g0.y = laddr;                                // lds_addr
    g0.z = (unsigned int)gaddr;                  // global_addr[31:0]
    g0.w = (unsigned int)(gaddr >> 32) | (2u << 30);  // addr[56:32] | type=2
    v8i g1;
    g1[0] = (int)((1u << 16)      // data_size = 2 bytes
                | (1u << 20)      // pad_enable
                | (3u << 22)      // pad_interval: 16 DWORDs
                | (3u << 25));    // pad_amount: 4 DWORDs
    g1[1] = (int)(2048u << 16);   // tensor_dim0[15:0] (=2048) in bits[63:48]
    g1[2] = (int)(2048u << 16);   // dim0 hi=0 | tensor_dim1[15:0] (=2048)
    g1[3] = (int)(32u << 16);     // dim1 hi=0 | tile_dim0 = 32
    g1[4] = (int)(256u);          // tile_dim1 = 256 | tile_dim2 = 0
    g1[5] = 2048;                 // tensor_dim0_stride low32
    g1[6] = 0;
    g1[7] = 0;
    v4i gz4 = {0, 0, 0, 0};
    v8i gz8 = {0, 0, 0, 0, 0, 0, 0, 0};
    __builtin_amdgcn_tensor_load_to_lds(g0, g1, gz4, gz4, gz8, 0);
  };

  auto stage_x = [&](int kc, int buf) {
    const bf16_t* src = X + (size_t)(kc * 32 + xkq * 4) * XCOLP + colBase + xc;
    union { bf16_t h[4]; uint2 u; } pk;
    pk.h[0] = src[0];
    pk.h[1] = src[XCOLP];
    pk.h[2] = src[2 * XCOLP];
    pk.h[3] = src[3 * XCOLP];
    *(uint2*)(&lxt[buf][xc * 40 + xkq * 4]) = pk.u;
  };

  // Prologue: fill buffer 0
  if (wave == 0) issue_tdm(0, 0);
  stage_x(0, 0);
  if (wave == 0) __builtin_amdgcn_s_wait_tensorcnt(0);
  __syncthreads();

  const int NCHUNK = Nq / 32;
  for (int kc = 0; kc < NCHUNK; ++kc) {
    int cur = kc & 1, nxt = cur ^ 1;
    bool haveNext = (kc + 1) < NCHUNK;

    bf16_t xr0 = (bf16_t)0.f, xr1 = xr0, xr2 = xr0, xr3 = xr0;
    if (haveNext) {
      if (wave == 0) issue_tdm(kc + 1, nxt);
      const bf16_t* src = X + (size_t)((kc + 1) * 32 + xkq * 4) * XCOLP + colBase + xc;
      xr0 = src[0];
      xr1 = src[XCOLP];
      xr2 = src[2 * XCOLP];
      xr3 = src[3 * XCOLP];
    }

    v16bf a0 = frag_a_rm(&la[cur][(wave * 32) * 40], 40);
    v16bf a1 = frag_a_rm(&la[cur][(wave * 32 + 16) * 40], 40);
    v16bf b0 = frag_b_cm(&lxt[cur][0], 40);
    v16bf b1 = frag_b_cm(&lxt[cur][16 * 40], 40);
    acc00 = wmma_bf16(a0, b0, acc00);
    acc01 = wmma_bf16(a0, b1, acc01);
    acc10 = wmma_bf16(a1, b0, acc10);
    acc11 = wmma_bf16(a1, b1, acc11);

    if (haveNext) {
      union { bf16_t h[4]; uint2 u; } pk;
      pk.h[0] = xr0; pk.h[1] = xr1; pk.h[2] = xr2; pk.h[3] = xr3;
      *(uint2*)(&lxt[nxt][xc * 40 + xkq * 4]) = pk.u;
      if (wave == 0) __builtin_amdgcn_s_wait_tensorcnt(0);
    }
    __syncthreads();
  }

  int lane = tid & 31;
  int colL = lane & 15;
  int rb   = (lane & 16) ? 8 : 0;
  v8f accs[2][2] = {{acc00, acc01}, {acc10, acc11}};
#pragma unroll
  for (int mt = 0; mt < 2; ++mt)
#pragma unroll
    for (int ct = 0; ct < 2; ++ct)
#pragma unroll
      for (int v = 0; v < 8; ++v) {
        int row = rowBase + wave * 32 + mt * 16 + rb + v;
        int col = colBase + ct * 16 + colL;
        AX[(size_t)row * XCOLP + col] = (bf16_t)accs[mt][ct][v];
      }
}

// ---- Gate: per-node [32 x 160] @ [160 x 128] + bias, sigmoid -> z, r -------
__global__ __launch_bounds__(256) void gate_mm(const bf16_t* __restrict__ X,
                                               const bf16_t* __restrict__ AX,
                                               const bf16_t* __restrict__ Wg,
                                               const float* __restrict__ bg,
                                               float* __restrict__ zbuf,
                                               float* __restrict__ rbuf) {
  int n  = blockIdx.x;
  int ot = threadIdx.x >> 5;
  v8f z8 = {0.f, 0.f, 0.f, 0.f, 0.f, 0.f, 0.f, 0.f};
  v8f acc0 = z8, acc1 = z8;
  const bf16_t* Wn = Wg + (size_t)n * (NKC * 8 * 16 * 32);

#pragma unroll
  for (int kc = 0; kc < NKC; ++kc) {
    v16bf b  = frag_b_cm(Wn + ((size_t)kc * 8 + ot) * (16 * 32), 32);
    v16bf a0 = frag_act(X, AX, n, 0, kc);
    v16bf a1 = frag_act(X, AX, n, 1, kc);
    acc0 = wmma_bf16(a0, b, acc0);
    acc1 = wmma_bf16(a1, b, acc1);
  }

  int lane = threadIdx.x & 31;
  int o    = ot * 16 + (lane & 15);
  int rb   = (lane & 16) ? 8 : 0;
  float bias = bg[n * OG + o];
  v8f accs[2] = {acc0, acc1};
#pragma unroll
  for (int mt = 0; mt < 2; ++mt)
#pragma unroll
    for (int v = 0; v < 8; ++v) {
      int b_ = mt * 16 + rb + v;
      float s = 1.f / (1.f + __expf(-(accs[mt][v] + bias)));
      size_t idx = ((size_t)b_ * Nq + n) * DOUTq;
      if (o < DOUTq) zbuf[idx + o] = s;
      else           rbuf[idx + (o - DOUTq)] = s;
    }
}

// ---- Candidate: [32 x 160] @ [160 x 64] + bias, tanh, GRU update -----------
__global__ __launch_bounds__(256) void cand_mm(const bf16_t* __restrict__ X,
                                               const bf16_t* __restrict__ AX,
                                               const bf16_t* __restrict__ Wc,
                                               const float* __restrict__ bc,
                                               const float* __restrict__ zbuf,
                                               float* __restrict__ hbuf,
                                               float* __restrict__ out,
                                               float* __restrict__ last, int t) {
  int n = blockIdx.x;
  int w = threadIdx.x >> 5;
  int ot = w & 3;
  int mt = w >> 2;
  v8f acc = {0.f, 0.f, 0.f, 0.f, 0.f, 0.f, 0.f, 0.f};
  const bf16_t* Wn = Wc + (size_t)n * (NKC * 4 * 16 * 32);

#pragma unroll
  for (int kc = 0; kc < NKC; ++kc) {
    v16bf b = frag_b_cm(Wn + ((size_t)kc * 4 + ot) * (16 * 32), 32);
    v16bf a = frag_act(X, AX, n, mt, kc);
    acc = wmma_bf16(a, b, acc);
  }

  int lane = threadIdx.x & 31;
  int o    = ot * 16 + (lane & 15);
  int rb   = (lane & 16) ? 8 : 0;
  float bias = bc[n * OC + o];
#pragma unroll
  for (int v = 0; v < 8; ++v) {
    int b_ = mt * 16 + rb + v;
    float hc = tanhf(acc[v] + bias);
    size_t idx = ((size_t)b_ * Nq + n) * DOUTq + o;
    float zv = zbuf[idx];
    float hp = hbuf[idx];
    float hn = zv * hp + (1.f - zv) * hc;
    hbuf[idx] = hn;
    out[(((size_t)b_ * Tq + t) * Nq + n) * DOUTq + o] = hn;
    if (t == Tq - 1) last[idx] = hn;
  }
}

// ---------------------------------------------------------------------------
extern "C" void kernel_launch(void* const* d_in, const int* in_sizes, int n_in,
                              void* d_out, int out_size, void* d_ws, size_t ws_size,
                              hipStream_t stream) {
  (void)in_sizes; (void)n_in; (void)out_size; (void)ws_size;
  const float* x    = (const float*)d_in[0];
  const float* h0   = (const float*)d_in[1];
  const float* E    = (const float*)d_in[2];
  const float* Wg0  = (const float*)d_in[3];
  const float* bg0  = (const float*)d_in[4];
  const float* Wc0  = (const float*)d_in[5];
  const float* bc0  = (const float*)d_in[6];

  float* out  = (float*)d_out;
  float* last = out + (size_t)Bq * Tq * Nq * DOUTq;

  char* p = (char*)d_ws;
  auto carve = [&](size_t bytes) {
    void* r = (void*)p;
    p += (bytes + 255) & ~(size_t)255;
    return r;
  };
  bf16_t* A    = (bf16_t*)carve((size_t)Nq * Nq * sizeof(bf16_t));
  bf16_t* Wg   = (bf16_t*)carve((size_t)Nq * NKC * 8 * 16 * 32 * sizeof(bf16_t));
  bf16_t* Wc   = (bf16_t*)carve((size_t)Nq * NKC * 4 * 16 * 32 * sizeof(bf16_t));
  float*  bg   = (float*) carve((size_t)Nq * OG * sizeof(float));
  float*  bc   = (float*) carve((size_t)Nq * OC * sizeof(float));
  float*  h    = (float*) carve((size_t)Bq * Nq * DOUTq * sizeof(float));
  bf16_t* X    = (bf16_t*)carve((size_t)Nq * XCOLP * sizeof(bf16_t));
  bf16_t* AX   = (bf16_t*)carve((size_t)Nq * XCOLP * sizeof(bf16_t));
  float*  zbuf = (float*) carve((size_t)Bq * Nq * DOUTq * sizeof(float));
  float*  rbuf = (float*) carve((size_t)Bq * Nq * DOUTq * sizeof(float));

  (void)hipMemcpyAsync(h, h0, (size_t)Bq * Nq * DOUTq * sizeof(float),
                       hipMemcpyDeviceToDevice, stream);

  build_adj<<<Nq, 256, 0, stream>>>(E, A);
  build_weights<<<Nq, 256, 0, stream>>>(E, Wg0, bg0, Wc0, bc0, Wg, Wc, bg, bc);

  const int packBlocks = (int)(((size_t)Nq * Bq * CP + 255) / 256);
  dim3 gemmGrid(Nq / 256, XCOLP / 32);

  for (int t = 0; t < Tq; ++t) {
    pack_gate<<<packBlocks, 256, 0, stream>>>(x, h, X, t);
    gemm_ax<<<gemmGrid, 256, 0, stream>>>(A, X, AX);
    gate_mm<<<Nq, 256, 0, stream>>>(X, AX, Wg, bg, zbuf, rbuf);
    pack_cand<<<packBlocks, 256, 0, stream>>>(x, h, rbuf, X, t);
    gemm_ax<<<gemmGrid, 256, 0, stream>>>(A, X, AX);
    cand_mm<<<Nq, 256, 0, stream>>>(X, AX, Wc, bc, zbuf, h, out, last, t);
  }
}
